// RNN_decoder_rate2_6270652252486
// MI455X (gfx1250) — matrix operand order repaired and checked by reading, not compiled, not run on hardware
//
#include <hip/hip_runtime.h>
#include <hip/hip_bf16.h>

typedef __attribute__((ext_vector_type(16))) _Float16 v16h;
typedef __attribute__((ext_vector_type(8)))  _Float16 v8h;
typedef __attribute__((ext_vector_type(8)))  float    v8f;

#define B_   1024
#define L_   200
#define H_   128
#define G3_  384      // 3*H
#define D1_  256      // layer-1 input dim (2H)
#define HSTRIDE 136   // padded LDS row stride (halves) for h tile

// fast activations: v_rcp_f32 + v_exp_f32 (keeps the serial recurrence path short)
__device__ __forceinline__ float fast_sigmoid(float x) {
  return __builtin_amdgcn_rcpf(1.0f + __expf(-x));
}
__device__ __forceinline__ float fast_tanh(float x) {
  // tanh(x) = 2*sigmoid(2x) - 1
  return fmaf(2.0f, __builtin_amdgcn_rcpf(1.0f + __expf(-2.0f * x)), -1.0f);
}

__device__ __forceinline__ v8f wmma16(v16h a, v16h b, v8f c) {
  return __builtin_amdgcn_wmma_f32_16x16x32_f16(false, a, false, b, (short)0, c, false, false);
}

// A fragment (16x32 f16): lane(0-15)=row m, halves: K = half*8..+7 and K = 16+half*8..+7
__device__ __forceinline__ v16h load_a_frag(const _Float16* base, int rowStride, int lane, int k0) {
  const int row = lane & 15, half = lane >> 4;
  const _Float16* p = base + (size_t)row * rowStride + k0 + half * 8;
  v8h lo = *(const v8h*)(p);
  v8h hi = *(const v8h*)(p + 16);
  return __builtin_shufflevector(lo, hi, 0,1,2,3,4,5,6,7,8,9,10,11,12,13,14,15);
}

// B fragment (32x16 f16) from row-major W[N][K]: lane col = n0+lane%16,
// halves: K = half*8..+7 and K = 16+half*8..+7
__device__ __forceinline__ v16h load_b_frag(const _Float16* base, int rowStride, int lane, int n0, int k0) {
  const int colr = lane & 15, half = lane >> 4;
  const _Float16* p = base + (size_t)(n0 + colr) * rowStride + k0 + half * 8;
  v8h lo = *(const v8h*)(p);
  v8h hi = *(const v8h*)(p + 16);
  return __builtin_shufflevector(lo, hi, 0,1,2,3,4,5,6,7,8,9,10,11,12,13,14,15);
}

// ---------------- small helper kernels ----------------

__global__ void cvt_f16_kernel(const float* __restrict__ src, _Float16* __restrict__ dst, int n) {
  int i = blockIdx.x * 256 + threadIdx.x;
  if (i < n) dst[i] = (_Float16)src[i];
}

// code[b][j] = b_lin[j] + sum_k received[b][k] * W_lin[j][k]
__global__ void dec_linear_kernel(const float* __restrict__ recv, const float* __restrict__ W,
                                  const float* __restrict__ bias, float* __restrict__ code) {
  int idx = blockIdx.x * 256 + threadIdx.x;
  if (idx >= B_ * 400) return;
  int b = idx / 400, j = idx - b * 400;
  const float4* x4 = (const float4*)(recv + (size_t)b * 400);
  const float4* w4 = (const float4*)(W + (size_t)j * 400);
  float acc = bias[j];
  #pragma unroll 4
  for (int k = 0; k < 100; ++k) {
    float4 xv = x4[k], wv = w4[k];
    acc = fmaf(xv.x, wv.x, acc);
    acc = fmaf(xv.y, wv.y, acc);
    acc = fmaf(xv.z, wv.z, acc);
    acc = fmaf(xv.w, wv.w, acc);
  }
  code[idx] = acc;
}

// out[b*L+l] = sigmoid(h1[b][l][:] . Wf + bf)
__global__ void final_proj_kernel(const _Float16* __restrict__ h1, const float* __restrict__ Wf,
                                  const float* __restrict__ bf, float* __restrict__ out) {
  int idx = blockIdx.x * 256 + threadIdx.x;
  if (idx >= B_ * L_) return;
  const v8h* hp = (const v8h*)(h1 + (size_t)idx * D1_);
  float acc = bf[0];
  #pragma unroll 4
  for (int c = 0; c < D1_ / 8; ++c) {
    v8h hv = hp[c];
    #pragma unroll
    for (int j = 0; j < 8; ++j) acc = fmaf((float)hv[j], Wf[c * 8 + j], acc);
  }
  out[idx] = fast_sigmoid(acc);
}

// ---------------- layer 0: bidirectional GRU, din = 2 ----------------
// grid (64, 2): x = batch tile of 16 rows, y = direction. 256 threads = 8 waves.
// Whh B-fragments live in registers for the whole 200-step loop; LDS holds only
// the ping-pong h tiles used to redistribute h into the A-fragment layout.
__global__ void gru_layer0_kernel(const float* __restrict__ code,       // [B][400]
                                  const _Float16* __restrict__ whh16,   // [2][384][128]
                                  const float* __restrict__ wih_f, const float* __restrict__ wih_b, // [384][2]
                                  const float* __restrict__ bih_f, const float* __restrict__ bhh_f,
                                  const float* __restrict__ bih_b, const float* __restrict__ bhh_b,
                                  _Float16* __restrict__ h0)            // [B][L][256]
{
  extern __shared__ char smem[];
  _Float16* hbufA = (_Float16*)smem;                   // 16*HSTRIDE halves (ping)
  _Float16* hbufB = hbufA + 16 * HSTRIDE;              // 16*HSTRIDE halves (pong)

  const int dir  = blockIdx.y;
  const int b0   = blockIdx.x * 16;
  const int tid  = threadIdx.x;
  const int lane = tid & 31, wave = tid >> 5;

  const _Float16* whhW = whh16 + (size_t)dir * (G3_ * H_);
  const float* wih = dir ? wih_b : wih_f;
  const float* bih = dir ? bih_b : bih_f;
  const float* bhh = dir ? bhh_b : bhh_f;

  // hoist all Whh B-fragments for this wave into registers (loop-invariant)
  v16h bwr[4], bwz[4], bwn[4];
  #pragma unroll
  for (int kt = 0; kt < 4; ++kt) {
    bwr[kt] = load_b_frag(whhW, H_, lane, 0,      kt * 32);
    bwz[kt] = load_b_frag(whhW, H_, lane, H_,     kt * 32);
    bwn[kt] = load_b_frag(whhW, H_, lane, 2 * H_, kt * 32);
  }

  for (int i = tid; i < 2 * 16 * HSTRIDE; i += 256) hbufA[i] = (_Float16)0.0f;

  const int col  = wave * 16 + (lane & 15);            // hidden column owned by this lane
  const int rrow = col, zrow = H_ + col, nrow = 2 * H_ + col;
  const float wr0 = wih[rrow*2], wr1 = wih[rrow*2+1];
  const float wz0 = wih[zrow*2], wz1 = wih[zrow*2+1];
  const float wn0 = wih[nrow*2], wn1 = wih[nrow*2+1];
  const float brz_r = bih[rrow] + bhh[rrow];
  const float brz_z = bih[zrow] + bhh[zrow];
  const float bi_n  = bih[nrow], bh_n = bhh[nrow];
  const int mbase = (lane >> 4) * 8;

  float hs[8];                                         // per-lane hidden state h[m][col]
  #pragma unroll
  for (int v = 0; v < 8; ++v) hs[v] = 0.0f;

  __syncthreads();

  _Float16* hr = hbufA;   // read buffer (state at step t)
  _Float16* hw = hbufB;   // write buffer (state at step t+1)

  for (int t = 0; t < L_; ++t) {
    const int l = dir ? (L_ - 1 - t) : t;

    v16h a[4];
    #pragma unroll
    for (int kt = 0; kt < 4; ++kt) a[kt] = load_a_frag(hr, HSTRIDE, lane, kt * 32);

    v8f cr = {}, cz = {}, cn = {};
    #pragma unroll
    for (int kt = 0; kt < 4; ++kt) {
      cr = wmma16(a[kt], bwr[kt], cr);
      cz = wmma16(a[kt], bwz[kt], cz);
      cn = wmma16(a[kt], bwn[kt], cn);
    }

    #pragma unroll
    for (int v = 0; v < 8; ++v) {
      const int m = mbase + v;
      const float x0 = code[(size_t)(b0 + m) * 400 + 2 * l];
      const float x1 = code[(size_t)(b0 + m) * 400 + 2 * l + 1];
      const float r = fast_sigmoid(fmaf(x0, wr0, fmaf(x1, wr1, cr[v] + brz_r)));
      const float z = fast_sigmoid(fmaf(x0, wz0, fmaf(x1, wz1, cz[v] + brz_z)));
      const float n = fast_tanh(fmaf(x0, wn0, fmaf(x1, wn1, bi_n)) + r * (cn[v] + bh_n));
      hs[v] = (1.0f - z) * n + z * hs[v];
      const _Float16 hv = (_Float16)hs[v];
      hw[m * HSTRIDE + col] = hv;
      h0[(size_t)(b0 + m) * L_ * D1_ + (size_t)l * D1_ + dir * H_ + col] = hv;
    }
    __syncthreads();
    _Float16* tmp = hr; hr = hw; hw = tmp;
  }
}

// ---------------- layer 1: bidirectional GRU, din = 256 ----------------
__global__ void gru_layer1_kernel(const _Float16* __restrict__ h0,      // [B][L][256]
                                  const _Float16* __restrict__ wih16,   // [2][384][256]
                                  const _Float16* __restrict__ whh16,   // [2][384][128]
                                  const float* __restrict__ bih_f, const float* __restrict__ bhh_f,
                                  const float* __restrict__ bih_b, const float* __restrict__ bhh_b,
                                  _Float16* __restrict__ h1)            // [B][L][256]
{
  extern __shared__ char smem[];
  _Float16* hbufA = (_Float16*)smem;
  _Float16* hbufB = hbufA + 16 * HSTRIDE;

  const int dir  = blockIdx.y;
  const int b0   = blockIdx.x * 16;
  const int tid  = threadIdx.x;
  const int lane = tid & 31, wave = tid >> 5;

  const _Float16* whhW = whh16 + (size_t)dir * (G3_ * H_);
  const _Float16* wih  = wih16 + (size_t)dir * (G3_ * D1_);
  const float* bih = dir ? bih_b : bih_f;
  const float* bhh = dir ? bhh_b : bhh_f;

  // loop-invariant Whh B-fragments in registers (96 VGPRs)
  v16h bwr[4], bwz[4], bwn[4];
  #pragma unroll
  for (int kt = 0; kt < 4; ++kt) {
    bwr[kt] = load_b_frag(whhW, H_, lane, 0,      kt * 32);
    bwz[kt] = load_b_frag(whhW, H_, lane, H_,     kt * 32);
    bwn[kt] = load_b_frag(whhW, H_, lane, 2 * H_, kt * 32);
  }

  for (int i = tid; i < 2 * 16 * HSTRIDE; i += 256) hbufA[i] = (_Float16)0.0f;

  const int col  = wave * 16 + (lane & 15);
  const int rrow = col, zrow = H_ + col, nrow = 2 * H_ + col;
  const float brz_r = bih[rrow] + bhh[rrow];
  const float brz_z = bih[zrow] + bhh[zrow];
  const float bi_n  = bih[nrow], bh_n = bhh[nrow];
  const int mbase = (lane >> 4) * 8;

  float hs[8];
  #pragma unroll
  for (int v = 0; v < 8; ++v) hs[v] = 0.0f;

  __syncthreads();

  _Float16* hr = hbufA;
  _Float16* hw = hbufB;

  for (int t = 0; t < L_; ++t) {
    const int l = dir ? (L_ - 1 - t) : t;
    const _Float16* xbase = h0 + (size_t)b0 * L_ * D1_ + (size_t)l * D1_;

    v8f cr = {}, cz = {}, cxn = {}, chn = {};
    // input part: [16 x 384] += x[16 x 256] @ Wih^T  (x and Wih streamed from L2)
    #pragma unroll
    for (int kt = 0; kt < 8; ++kt) {
      v16h a = load_a_frag(xbase, L_ * D1_, lane, kt * 32);
      cr  = wmma16(a, load_b_frag(wih, D1_, lane, 0,      kt * 32), cr);
      cz  = wmma16(a, load_b_frag(wih, D1_, lane, H_,     kt * 32), cz);
      cxn = wmma16(a, load_b_frag(wih, D1_, lane, 2 * H_, kt * 32), cxn);
    }
    // recurrent part: += h[16 x 128] @ Whh^T  (Whh in registers, h from LDS)
    v16h ah[4];
    #pragma unroll
    for (int kt = 0; kt < 4; ++kt) ah[kt] = load_a_frag(hr, HSTRIDE, lane, kt * 32);
    #pragma unroll
    for (int kt = 0; kt < 4; ++kt) {
      cr  = wmma16(ah[kt], bwr[kt], cr);
      cz  = wmma16(ah[kt], bwz[kt], cz);
      chn = wmma16(ah[kt], bwn[kt], chn);
    }

    #pragma unroll
    for (int v = 0; v < 8; ++v) {
      const int m = mbase + v;
      const float r = fast_sigmoid(cr[v] + brz_r);
      const float z = fast_sigmoid(cz[v] + brz_z);
      const float n = fast_tanh(cxn[v] + bi_n + r * (chn[v] + bh_n));
      hs[v] = (1.0f - z) * n + z * hs[v];
      const _Float16 hv = (_Float16)hs[v];
      hw[m * HSTRIDE + col] = hv;
      h1[(size_t)(b0 + m) * L_ * D1_ + (size_t)l * D1_ + dir * H_ + col] = hv;
    }
    __syncthreads();
    _Float16* tmp = hr; hr = hw; hw = tmp;
  }
}

// ---------------- launch ----------------

extern "C" void kernel_launch(void* const* d_in, const int* in_sizes, int n_in,
                              void* d_out, int out_size, void* d_ws, size_t ws_size,
                              hipStream_t stream) {
  const float* recv  = (const float*)d_in[0];
  const float* W_lin = (const float*)d_in[1];
  const float* b_lin = (const float*)d_in[2];
  const float* Wih0f = (const float*)d_in[3];
  const float* Whh0f = (const float*)d_in[4];
  const float* bih0f = (const float*)d_in[5];
  const float* bhh0f = (const float*)d_in[6];
  const float* Wih0b = (const float*)d_in[7];
  const float* Whh0b = (const float*)d_in[8];
  const float* bih0b = (const float*)d_in[9];
  const float* bhh0b = (const float*)d_in[10];
  const float* Wih1f = (const float*)d_in[11];
  const float* Whh1f = (const float*)d_in[12];
  const float* bih1f = (const float*)d_in[13];
  const float* bhh1f = (const float*)d_in[14];
  const float* Wih1b = (const float*)d_in[15];
  const float* Whh1b = (const float*)d_in[16];
  const float* bih1b = (const float*)d_in[17];
  const float* bhh1b = (const float*)d_in[18];
  const float* Wf    = (const float*)d_in[19];
  const float* bf    = (const float*)d_in[20];
  float* out = (float*)d_out;

  char* wsb = (char*)d_ws;
  size_t off = 0;
  float*     code = (float*)(wsb + off);     off += (size_t)B_ * 400 * sizeof(float);
  _Float16*  h0   = (_Float16*)(wsb + off);  off += (size_t)B_ * L_ * D1_ * sizeof(_Float16);
  _Float16*  h1   = (_Float16*)(wsb + off);  off += (size_t)B_ * L_ * D1_ * sizeof(_Float16);
  _Float16*  whh0 = (_Float16*)(wsb + off);  off += (size_t)2 * G3_ * H_ * sizeof(_Float16);
  _Float16*  whh1 = (_Float16*)(wsb + off);  off += (size_t)2 * G3_ * H_ * sizeof(_Float16);
  _Float16*  wih1 = (_Float16*)(wsb + off);  off += (size_t)2 * G3_ * D1_ * sizeof(_Float16);

  const int nWhh = G3_ * H_;   // 49152
  const int nWih = G3_ * D1_;  // 98304

  // weight conversion f32 -> f16
  cvt_f16_kernel<<<(nWhh + 255) / 256, 256, 0, stream>>>(Whh0f, whh0,        nWhh);
  cvt_f16_kernel<<<(nWhh + 255) / 256, 256, 0, stream>>>(Whh0b, whh0 + nWhh, nWhh);
  cvt_f16_kernel<<<(nWhh + 255) / 256, 256, 0, stream>>>(Whh1f, whh1,        nWhh);
  cvt_f16_kernel<<<(nWhh + 255) / 256, 256, 0, stream>>>(Whh1b, whh1 + nWhh, nWhh);
  cvt_f16_kernel<<<(nWih + 255) / 256, 256, 0, stream>>>(Wih1f, wih1,        nWih);
  cvt_f16_kernel<<<(nWih + 255) / 256, 256, 0, stream>>>(Wih1b, wih1 + nWih, nWih);

  // dec_linear
  dec_linear_kernel<<<(B_ * 400 + 255) / 256, 256, 0, stream>>>(recv, W_lin, b_lin, code);

  const size_t lds_bytes = (size_t)2 * 16 * HSTRIDE * sizeof(_Float16);  // 8704 B

  dim3 gruGrid(B_ / 16, 2);
  gru_layer0_kernel<<<gruGrid, 256, lds_bytes, stream>>>(
      code, whh0, Wih0f, Wih0b, bih0f, bhh0f, bih0b, bhh0b, h0);

  gru_layer1_kernel<<<gruGrid, 256, lds_bytes, stream>>>(
      h0, wih1, whh1, bih1f, bhh1f, bih1b, bhh1b, h1);

  final_proj_kernel<<<(B_ * L_ + 255) / 256, 256, 0, stream>>>(h1, Wf, bf, out);
}
